// ShiftedWindowAttention_50319836840387
// MI455X (gfx1250) — matrix operand order repaired and checked by reading, not compiled, not run on hardware
//
#include <hip/hip_runtime.h>
#include <hip/hip_bf16.h>

typedef __attribute__((ext_vector_type(16))) _Float16 v16h;
typedef __attribute__((ext_vector_type(8)))  float    v8f;

#define C_     96
#define H_     512
#define W_     512
#define WS_    8
#define N_     64        // tokens per window
#define HEADS_ 6
#define HD_    16
#define SHIFT_ 4
#define NWX    64        // windows per row
#define HW_    (H_ * W_)
#define WC_    (W_ * C_)
#define QKVLD  (3 * C_)  // 288
#define XLD    98        // padded row stride for xbuf (odd dword stride -> no LDS bank conflicts)

// ---- WMMA fragment loaders (contiguous k order, same in A and B, wave32) ----

// A: 16x32 f16 from row-major [m][k] source (LDS). kvalid zero-pads K.
// lanes 0-15: M=lane, K = 0..15; lanes 16-31: M=lane-16, K = 16..31
__device__ inline v16h lda_rm(const _Float16* src, int m0, int k0, int ld, int kvalid) {
  const int lane = threadIdx.x & 31;
  const int m = m0 + (lane & 15);
  const int kb = (lane >> 4) << 4;           // 0 or 16
  v16h a;
#pragma unroll
  for (int i = 0; i < 8; ++i) {
    int kr = kb + i;
    a[i] = (kr < kvalid) ? src[m * ld + k0 + kr] : (_Float16)0.f;
  }
#pragma unroll
  for (int i = 0; i < 8; ++i) {
    int kr = kb + 8 + i;
    a[8 + i] = (kr < kvalid) ? src[m * ld + k0 + kr] : (_Float16)0.f;
  }
  return a;
}

// B: 32x16 f16 where source is row-major [n][k] (i.e. B[k][n] = src[n][k]).
// lanes 0-15: N=lane, K=0..15; lanes 16-31: N=lane-16, K=16..31
__device__ inline v16h ldb_nk_h(const _Float16* src, int n0, int k0, int ld, int kvalid) {
  const int lane = threadIdx.x & 31;
  const int n = n0 + (lane & 15);
  const int kb = (lane >> 4) << 4;           // 0 or 16
  v16h b;
#pragma unroll
  for (int i = 0; i < 16; ++i) {
    int kr = kb + i;
    b[i] = (kr < kvalid) ? src[n * ld + k0 + kr] : (_Float16)0.f;
  }
  return b;
}

// Same but f32 global source (weights), converted to f16 on the fly.
// Per lane this reads 16 consecutive floats (one weight row chunk) -> b128 loads.
__device__ inline v16h ldb_nk_f32(const float* __restrict__ src, int n0, int k0, int ld, int kvalid) {
  const int lane = threadIdx.x & 31;
  const int n = n0 + (lane & 15);
  const int kb = (lane >> 4) << 4;
  v16h b;
#pragma unroll
  for (int i = 0; i < 16; ++i) {
    int kr = kb + i;
    b[i] = (kr < kvalid) ? (_Float16)src[n * ld + k0 + kr] : (_Float16)0.f;
  }
  return b;
}

// B: 32x16 f16 where source is row-major [k][n] (V matrix: B[k][n] = src[k][n]).
__device__ inline v16h ldb_kn_h(const _Float16* src, int k0, int n0, int ld) {
  const int lane = threadIdx.x & 31;
  const int n = n0 + (lane & 15);
  const int kb = (lane >> 4) << 4;
  v16h b;
#pragma unroll
  for (int i = 0; i < 16; ++i)
    b[i] = src[(k0 + kb + i) * ld + n];
  return b;
}

// shifted-window region code for the attention mask (rows and cols symmetric)
__device__ inline int reg_code(int g) {
  return (g < (H_ - WS_)) ? 0 : ((g < (H_ - SHIFT_)) ? 1 : 2);
}

__global__ __launch_bounds__(256, 1)
void swin_attn_fused(const float* __restrict__ x,
                     const float* __restrict__ bias_tbl,   // (225, 6)
                     const float* __restrict__ w_qkv,      // (288, 96)
                     const float* __restrict__ b_qkv,      // (288,)
                     const float* __restrict__ w_proj,     // (96, 96)
                     const float* __restrict__ b_proj,     // (96,)
                     float* __restrict__ out)              // (1, 96, 512, 512)
{
  __shared__ _Float16 xbuf[N_ * XLD];      // window input -> reused as attn-out (12.25 KB)
  __shared__ _Float16 qkvs[N_ * QKVLD];    // q|k|v, f16, token-major (36 KB)
  __shared__ _Float16 sp[N_ * N_];         // logits -> probs in place (8 KB)

  const int tid  = threadIdx.x;
  const int lane = tid & 31;
  const int wave = tid >> 5;
  const int win  = blockIdx.x;
  const int wy = win / NWX, wx = win - wy * NWX;
  const int hbase = wy * WS_, wbase = wx * WS_;

  // warm weights into cache (global_prefetch_b8); tiny vs 192MB L2
  if (tid < 128) {
    __builtin_prefetch(w_qkv + tid * 216, 0, 3);
    __builtin_prefetch(w_proj + tid * 72, 0, 3);
  }

  // ---- Phase 1: gather shifted window, fp32 -> f16 --------------------
  // Reference does x.reshape(b,h,w,c) as a RAW reshape of (b,c,h,w) memory,
  // rolls by -shift on that view, reshapes back: a flat-index permutation
  //   f -> (h1,w1,c1) in the (h,w,c) view -> rolled source flat index g.
  // Iterate idx = c*64 + t so consecutive lanes step j (innermost window
  // col): the permutation is piecewise-contiguous there -> 32B coalesced
  // runs instead of isolated dwords. XLD=98 keeps the LDS store conflict-free.
  for (int idx = tid; idx < N_ * C_; idx += 256) {
    int c = idx >> 6, t = idx & 63;
    int i = t >> 3, j = t & 7;
    int f  = c * HW_ + (hbase + i) * W_ + (wbase + j);
    int h1 = f / WC_;  int r0 = f - h1 * WC_;
    int w1 = r0 / C_;  int c1 = r0 - w1 * C_;
    int h2 = h1 + SHIFT_; if (h2 >= H_) h2 -= H_;
    int w2 = w1 + SHIFT_; if (w2 >= W_) w2 -= W_;
    xbuf[t * XLD + c] = (_Float16)x[h2 * WC_ + w2 * C_ + c1];
  }
  __syncthreads();

  // ---- Phase 2: qkv = Xw @ Wqkv^T + b  (q pre-scaled by 0.25) ---------
  // 4 M-tiles x 18 N-tiles, K=96 -> 3 WMMA steps; 9 tiles per wave
  for (int tile = wave; tile < 72; tile += 8) {
    int mt = tile & 3, nt = tile >> 2;
    int m0 = mt * 16, n0 = nt * 16;
    v8f acc = {};
#pragma unroll
    for (int kk = 0; kk < C_; kk += 32) {
      v16h a = lda_rm(xbuf, m0, kk, XLD, 32);
      v16h b = ldb_nk_f32(w_qkv, n0, kk, C_, 32);
      acc = __builtin_amdgcn_wmma_f32_16x16x32_f16(false, a, false, b,
                                                   (short)0, acc, false, false);
    }
    int n = n0 + (lane & 15);
    float bq = b_qkv[n];
    float sc = (n < C_) ? 0.25f : 1.0f;     // scale = hd^-0.5, applied to q only
    int moff = (lane >> 4) << 3;
#pragma unroll
    for (int r = 0; r < 8; ++r) {
      int m = m0 + moff + r;
      qkvs[m * QKVLD + n] = (_Float16)((acc[r] + bq) * sc);
    }
  }
  __syncthreads();

  // ---- Phase 3: per-head attention ------------------------------------
  for (int h = 0; h < HEADS_; ++h) {
    // S = q @ k^T (K=16 zero-padded to 32), + rel-pos bias + shift mask
    for (int tile = wave * 2; tile < wave * 2 + 2; ++tile) {
      int mt = tile >> 2, nt = tile & 3;
      int m0 = mt * 16, n0 = nt * 16;
      v16h a = lda_rm(qkvs, m0, h * HD_, QKVLD, HD_);
      v16h b = ldb_nk_h(qkvs, n0, C_ + h * HD_, QKVLD, HD_);
      v8f acc = {};
      acc = __builtin_amdgcn_wmma_f32_16x16x32_f16(false, a, false, b,
                                                   (short)0, acc, false, false);
      int n  = n0 + (lane & 15);
      int i2 = n >> 3, j2 = n & 7;
      int cnt2 = reg_code(hbase + i2) * 3 + reg_code(wbase + j2);
      int moff = (lane >> 4) << 3;
#pragma unroll
      for (int r = 0; r < 8; ++r) {
        int m = m0 + moff + r;
        int i1 = m >> 3, j1 = m & 7;
        int cnt1 = reg_code(hbase + i1) * 3 + reg_code(wbase + j1);
        int ridx = (i1 - i2 + 7) * 15 + (j1 - j2 + 7);      // (2ws-1) grid
        float v = acc[r] + bias_tbl[ridx * HEADS_ + h];
        if (cnt1 != cnt2) v -= 100.0f;
        sp[m * N_ + n] = (_Float16)v;
      }
    }
    __syncthreads();

    // row softmax in place (64 rows, one thread each)
    if (tid < N_) {
      _Float16* row = sp + tid * N_;
      float mx = -1e30f;
      for (int j = 0; j < N_; ++j) mx = fmaxf(mx, (float)row[j]);
      float s = 0.f;
      for (int j = 0; j < N_; ++j) {
        float e = __expf((float)row[j] - mx);
        row[j] = (_Float16)e;
        s += e;
      }
      float inv = 1.0f / s;
      for (int j = 0; j < N_; ++j) row[j] = (_Float16)((float)row[j] * inv);
    }
    __syncthreads();

    // O_h = P @ V_h : 4 M-tiles x 1 N-tile, K=64 -> 2 WMMA steps
    if (wave < 4) {
      int m0 = wave * 16;
      v8f acc = {};
#pragma unroll
      for (int kk = 0; kk < N_; kk += 32) {
        v16h a = lda_rm(sp, m0, kk, N_, 32);
        v16h b = ldb_kn_h(qkvs, kk, 2 * C_ + h * HD_, QKVLD);
        acc = __builtin_amdgcn_wmma_f32_16x16x32_f16(false, a, false, b,
                                                     (short)0, acc, false, false);
      }
      int n = lane & 15;
      int moff = (lane >> 4) << 3;
#pragma unroll
      for (int r = 0; r < 8; ++r)
        xbuf[(m0 + moff + r) * XLD + h * HD_ + n] = (_Float16)acc[r];  // xbuf reused
    }
    __syncthreads();   // sp/xbuf hazards before next head
  }

  // ---- Phase 4: proj + scatter to (c, h, w) ---------------------------
  // Y = O @ Wproj^T + b ; 4x6 tiles, K=96 -> 3 WMMA steps; 3 tiles per wave
  for (int tile = wave * 3; tile < wave * 3 + 3; ++tile) {
    int mt = tile & 3, nt = tile >> 2;
    int m0 = mt * 16, n0 = nt * 16;
    v8f acc = {};
#pragma unroll
    for (int kk = 0; kk < C_; kk += 32) {
      v16h a = lda_rm(xbuf, m0, kk, XLD, 32);
      v16h b = ldb_nk_f32(w_proj, n0, kk, C_, 32);
      acc = __builtin_amdgcn_wmma_f32_16x16x32_f16(false, a, false, b,
                                                   (short)0, acc, false, false);
    }
    int n = n0 + (lane & 15);     // output channel
    float bp = b_proj[n];
    int moff = (lane >> 4) << 3;
#pragma unroll
    for (int r = 0; r < 8; ++r) {
      int m = m0 + moff + r;
      int i = m >> 3, j = m & 7;
      // window_reverse -> (B, C, H, W); reference applies no reverse roll.
      // Per lane, r=0..7 hits 8 consecutive floats -> b128-fusable stores.
      out[n * HW_ + (hbase + i) * W_ + (wbase + j)] = acc[r] + bp;
    }
  }
}

extern "C" void kernel_launch(void* const* d_in, const int* in_sizes, int n_in,
                              void* d_out, int out_size, void* d_ws, size_t ws_size,
                              hipStream_t stream) {
  const float* x     = (const float*)d_in[0];
  const float* tbl   = (const float*)d_in[1];
  const float* wqkv  = (const float*)d_in[2];
  const float* bqkv  = (const float*)d_in[3];
  const float* wproj = (const float*)d_in[4];
  const float* bproj = (const float*)d_in[5];
  float* out = (float*)d_out;
  (void)in_sizes; (void)n_in; (void)out_size; (void)d_ws; (void)ws_size;

  dim3 grid(NWX * NWX);   // 4096 windows, one per workgroup (8 wave32)
  swin_attn_fused<<<grid, 256, 0, stream>>>(x, tbl, wqkv, bqkv, wproj, bproj, out);
}
